// CapyNet_33139967656492
// MI455X (gfx1250) — 8-bit matrix operand order repaired and checked by reading, compiled, not run on hardware
//
#include <hip/hip_runtime.h>
#include <hip/hip_bf16.h>
#include <math.h>

typedef __attribute__((ext_vector_type(8))) int   v8i;
typedef __attribute__((ext_vector_type(8))) float v8f;
typedef __attribute__((ext_vector_type(2))) float v2f;

__global__ void k_zero_f32(float* p, size_t n) {
  size_t t = (size_t)blockIdx.x * blockDim.x + threadIdx.x;
  if (t < n) p[t] = 0.f;
}
__global__ void k_copy_f32(const float* a, float* b, size_t n) {
  size_t t = (size_t)blockIdx.x * blockDim.x + threadIdx.x;
  if (t < n) b[t] = a[t];
}

__global__ void k_build_edges(const int* __restrict__ ei, int E, float* __restrict__ A, int N) {
  int e = blockIdx.x * blockDim.x + threadIdx.x;
  if (e < E) {
    int s = ei[e], d = ei[E + e];
    atomicAdd(&A[(size_t)s * N + d], 1.0f);
  }
}

__global__ void k_colsum_part(const float* __restrict__ A, int n, int slices, float* __restrict__ deg) {
  int t = blockIdx.x * blockDim.x + threadIdx.x;
  if (t >= n * slices) return;
  int s = t / n, j = t % n;
  int i0 = (int)(((long long)s * n) / slices);
  int i1 = (int)(((long long)(s + 1) * n) / slices);
  float acc = 0.f;
  for (int i = i0; i < i1; ++i) acc += A[(size_t)i * n + j];
  atomicAdd(&deg[j], acc);
}
__global__ void k_finalize_dis(const float* __restrict__ A, int n, const float* __restrict__ deg,
                               float* __restrict__ dis) {
  int j = blockIdx.x * blockDim.x + threadIdx.x;
  if (j >= n) return;
  float d = A[(size_t)j * n + j];
  float s = deg[j] + (d > 0.f ? 0.f : 2.f);
  dis[j] = (s > 0.f) ? rsqrtf(s) : 0.f;
}

__global__ void k_msg(const float* __restrict__ x, int Cin, const float* __restrict__ W, int Cout,
                      const float* __restrict__ dis, int n, float* __restrict__ msg) {
  int t = blockIdx.x * blockDim.x + threadIdx.x;
  if (t >= n * Cout) return;
  int i = t / Cout, h = t % Cout;
  float acc = 0.f;
  for (int c = 0; c < Cin; ++c) acc = fmaf(x[(size_t)i * Cin + c], W[c * Cout + h], acc);
  msg[(size_t)i * 32 + h] = acc * dis[i];
}

template <int COUT, bool RELU>
__global__ void k_aggregate(const float* __restrict__ A, int n, const float* __restrict__ msg,
                            const float* __restrict__ dis, const float* __restrict__ b,
                            float* __restrict__ out) {
  __shared__ __align__(16) float smsg[64 * 32];
  __shared__ float red[8 * 32 * COUT];
  const int tid = threadIdx.x;
  const int lane = tid & 31, w = tid >> 5;
  const int j0 = blockIdx.x * 32;
  const int jg = j0 + lane;
  const bool jok = jg < n;
  float acc[COUT];
#pragma unroll
  for (int h = 0; h < COUT; ++h) acc[h] = 0.f;
  for (int i0 = 0; i0 < n; i0 += 64) {
    if (i0 + 64 <= n) {
      const float4* m4 = (const float4*)(msg + (size_t)i0 * 32);
      float4* s4 = (float4*)smsg;
      s4[tid] = m4[tid];
      s4[tid + 256] = m4[tid + 256];
    } else {
      for (int d = tid; d < 2048; d += 256) {
        int ii = i0 + (d >> 5);
        smsg[d] = (ii < n) ? msg[(size_t)ii * 32 + (d & 31)] : 0.f;
      }
    }
    __syncthreads();
    int iend = (n - i0 < 64) ? (n - i0) : 64;
    for (int ii = w; ii < iend; ii += 8) {
      int i = i0 + ii;
      float a = jok ? A[(size_t)i * n + jg] : 0.f;
      if (i == jg && a <= 0.f) a += 2.f;
#pragma unroll
      for (int h = 0; h < COUT; ++h) acc[h] = fmaf(a, smsg[ii * 32 + h], acc[h]);
    }
    __syncthreads();
  }
#pragma unroll
  for (int h = 0; h < COUT; ++h) red[(w * 32 + lane) * COUT + h] = acc[h];
  __syncthreads();
  for (int o = tid; o < 32 * COUT; o += 256) {
    int jl = o / COUT, h = o % COUT;
    float s = 0.f;
#pragma unroll
    for (int wv = 0; wv < 8; ++wv) s += red[(wv * 32 + jl) * COUT + h];
    int jj = j0 + jl;
    if (jj < n) {
      float v = dis[jj] * s + b[h];
      if (RELU) v = fmaxf(v, 0.f);
      out[(size_t)jj * COUT + h] = v;
    }
  }
}

__global__ void k_u8pad(const float* __restrict__ Am, int n, int np,
                        unsigned char* __restrict__ U8, unsigned char* __restrict__ U8t) {
  __shared__ unsigned char t[32][33];
  int bx = blockIdx.x, by = blockIdx.y;
  int tx = threadIdx.x, ty = threadIdx.y;
  for (int r = ty; r < 32; r += 8) {
    int i = by * 32 + r, j = bx * 32 + tx;
    unsigned char v = 0;
    if (i < n && j < n) v = (unsigned char)(Am[(size_t)i * n + j] + (i == j ? 1.5f : 0.5f));
    t[r][tx] = v;
    U8[(size_t)i * np + j] = v;
  }
  __syncthreads();
  for (int r = ty; r < 32; r += 8)
    U8t[(size_t)(bx * 32 + r) * np + by * 32 + tx] = t[tx][r];
}

__global__ void k_f32pad(const float* __restrict__ Am, int n, int np, float* __restrict__ out) {
  size_t t = (size_t)blockIdx.x * blockDim.x + threadIdx.x;
  if (t >= (size_t)np * np) return;
  int i = (int)(t / np), j = (int)(t % np);
  out[t] = (i < n && j < n) ? (Am[(size_t)i * n + j] + (i == j ? 1.f : 0.f)) : 0.f;
}

__global__ void k_gemm_iu8(const unsigned char* __restrict__ U8, const unsigned char* __restrict__ U8t,
                           int n, int np, float* __restrict__ out) {
  __shared__ __align__(16) unsigned int sA[2][2048];
  __shared__ __align__(16) unsigned int sB[2][1024];
  const int tid = threadIdx.x;
  const int lane = tid & 31, w = tid >> 5;
  const int Mt = w & 3, Nt = w >> 2;
  const int half = lane >> 4, ml = lane & 15;
  const int gm0 = blockIdx.y * 64, gn0 = blockIdx.x * 32;
  const int row = tid >> 3, seg = tid & 7;
  const unsigned char* baseA = U8 + (size_t)gm0 * np;
  const unsigned char* baseB = U8t + (size_t)gn0 * np;

  v8i c;
#pragma unroll
  for (int r = 0; r < 8; ++r) c[r] = 0;

  uint4 va0, va1, vb0;
  auto gload = [&](int k0) {
    va0 = ((const uint4*)(baseA + (size_t)row * np + k0))[seg];
    va1 = ((const uint4*)(baseA + (size_t)(row + 32) * np + k0))[seg];
    vb0 = ((const uint4*)(baseB + (size_t)row * np + k0))[seg];
  };
  auto lstore = [&](int buf) {
    ((uint4*)sA[buf])[tid] = va0;
    ((uint4*)sA[buf])[tid + 256] = va1;
    ((uint4*)sB[buf])[tid] = vb0;
  };
  auto compute = [&](int buf) {
#pragma unroll
    for (int kc = 0; kc < 2; ++kc) {
      v8i a, b;
#pragma unroll
      for (int v = 0; v < 8; ++v) {
        int ka = ((v >> 2) << 3) + (((v >> 1) & 1) << 2) + (half << 1) + (v & 1);
        a[v] = (int)sA[buf][(Mt * 16 + ml) * 32 + kc * 16 + ka];
        int kb = ((v >> 2) << 3) + (((v >> 1) & 1) << 2) + (half << 1) + (v & 1);
        b[v] = (int)sB[buf][(Nt * 16 + ml) * 32 + kc * 16 + kb];
      }
      c = __builtin_amdgcn_wmma_i32_16x16x64_iu8(false, a, false, b, c, false, false);
    }
  };

  const int nch = np >> 7;
  gload(0);
  lstore(0);
  __syncthreads();
  for (int ci = 0; ci < nch; ++ci) {
    if (ci + 1 < nch) gload((ci + 1) << 7);
    compute(ci & 1);
    if (ci + 1 < nch) lstore((ci + 1) & 1);
    __syncthreads();
  }

#pragma unroll
  for (int r = 0; r < 8; ++r) {
    int gm = gm0 + Mt * 16 + r + (half << 3);
    int gn = gn0 + Nt * 16 + ml;
    if (gm < n && gn < n)
      out[(size_t)gm * n + gn] = (gm == gn) ? 0.f : (float)c[r];
  }
}

__global__ void k_gemm_f32(const float* __restrict__ Af, int n, int np, float* __restrict__ out) {
  __shared__ __align__(16) float sA[64 * 32];
  __shared__ __align__(16) float sB[32 * 32];
  const int tid = threadIdx.x;
  const int lane = tid & 31, w = tid >> 5;
  const int Mt = w & 3, Nt = w >> 2;
  const int half = lane >> 4, ml = lane & 15;
  const int gm0 = blockIdx.y * 64, gn0 = blockIdx.x * 32;
  const int row = tid >> 3, seg = tid & 7;
  v8f c;
#pragma unroll
  for (int r = 0; r < 8; ++r) c[r] = 0.f;
  for (int k0 = 0; k0 < np; k0 += 32) {
    float4 a0 = ((const float4*)(Af + (size_t)(gm0 + row) * np + k0))[seg];
    float4 a1 = ((const float4*)(Af + (size_t)(gm0 + row + 32) * np + k0))[seg];
    float4 b0 = ((const float4*)(Af + (size_t)(k0 + row) * np + gn0))[seg];
    __syncthreads();
    ((float4*)sA)[tid] = a0;
    ((float4*)sA)[tid + 256] = a1;
    ((float4*)sB)[tid] = b0;
    __syncthreads();
#pragma unroll
    for (int ks = 0; ks < 8; ++ks) {
      v2f a, b;
#pragma unroll
      for (int v = 0; v < 2; ++v) {
        a[v] = sA[(Mt * 16 + ml) * 32 + ks * 4 + half * 2 + v];
        b[v] = sB[(ks * 4 + half * 2 + v) * 32 + Nt * 16 + ml];
      }
      c = __builtin_amdgcn_wmma_f32_16x16x4_f32(false, a, false, b, (short)0, c, false, false);
    }
  }
#pragma unroll
  for (int r = 0; r < 8; ++r) {
    int gm = gm0 + Mt * 16 + r + (half << 3);
    int gn = gn0 + Nt * 16 + ml;
    if (gm < n && gn < n)
      out[(size_t)gm * n + gn] = (gm == gn) ? 0.f : c[r];
  }
}

__global__ void k_score(const float* __restrict__ x, const float* __restrict__ p, int n,
                        float* __restrict__ sc) {
  int i = blockIdx.x * blockDim.x + threadIdx.x;
  if (i >= n) return;
  float pn = 0.f, dot = 0.f;
#pragma unroll
  for (int h = 0; h < 32; ++h) pn = fmaf(p[h], p[h], pn);
  pn = sqrtf(pn);
#pragma unroll
  for (int h = 0; h < 32; ++h) dot = fmaf(x[(size_t)i * 32 + h], p[h], dot);
  sc[i] = tanhf(dot / pn);
}

__global__ void k_sort_topk(const float* __restrict__ sc, int n, int npad, int k,
                            int* __restrict__ perm, float* __restrict__ vals) {
  __shared__ float skey[4096];
  __shared__ int sidx[4096];
  for (int i = threadIdx.x; i < npad; i += blockDim.x) {
    skey[i] = (i < n) ? sc[i] : -INFINITY;
    sidx[i] = i;
  }
  __syncthreads();
  for (int size = 2; size <= npad; size <<= 1) {
    for (int stride = size >> 1; stride > 0; stride >>= 1) {
      for (int i = threadIdx.x; i < npad; i += blockDim.x) {
        int j = i ^ stride;
        if (j > i) {
          float ka = skey[i], kb = skey[j];
          int ia = sidx[i], ib = sidx[j];
          bool aFirst = (ka > kb) || (ka == kb && ia < ib);
          bool descBlock = ((i & size) == 0);
          if (descBlock ? !aFirst : aFirst) {
            skey[i] = kb; skey[j] = ka;
            sidx[i] = ib; sidx[j] = ia;
          }
        }
      }
      __syncthreads();
    }
  }
  for (int i = threadIdx.x; i < k; i += blockDim.x) {
    perm[i] = sidx[i];
    vals[i] = skey[i];
  }
}

__global__ void k_gather_x(const float* __restrict__ x, const int* __restrict__ perm,
                           const float* __restrict__ vals, int k, float* __restrict__ out) {
  int t = blockIdx.x * blockDim.x + threadIdx.x;
  if (t >= k * 32) return;
  int r = t >> 5, h = t & 31;
  out[t] = x[(size_t)perm[r] * 32 + h] * vals[r];
}
__global__ void k_gather_A(const float* __restrict__ Asq, int n, const int* __restrict__ perm, int k,
                           float* __restrict__ out) {
  size_t t = (size_t)blockIdx.x * blockDim.x + threadIdx.x;
  if (t >= (size_t)k * k) return;
  int r = (int)(t / k), c = (int)(t % k);
  out[t] = Asq[(size_t)perm[r] * n + perm[c]];
}
__global__ void k_scatter_add(float* __restrict__ xt, const int* __restrict__ perm,
                              const float* __restrict__ xc, int k) {
  int t = blockIdx.x * blockDim.x + threadIdx.x;
  if (t >= k * 32) return;
  int r = t >> 5, h = t & 31;
  xt[(size_t)perm[r] * 32 + h] += xc[t];
}

__global__ void k_logsoftmax3(const float* __restrict__ z, int n, float* __restrict__ out) {
  int i = blockIdx.x * blockDim.x + threadIdx.x;
  if (i >= n) return;
  float a = z[i * 3 + 0], b = z[i * 3 + 1], c = z[i * 3 + 2];
  float m = fmaxf(a, fmaxf(b, c));
  float l = logf(expf(a - m) + expf(b - m) + expf(c - m));
  out[i * 3 + 0] = a - m - l;
  out[i * 3 + 1] = b - m - l;
  out[i * 3 + 2] = c - m - l;
}

extern "C" void kernel_launch(void* const* d_in, const int* in_sizes, int n_in,
                              void* d_out, int out_size, void* d_ws, size_t ws_size,
                              hipStream_t stream) {
  (void)n_in; (void)out_size; (void)ws_size;
  const float* x_in  = (const float*)d_in[0];
  const int*   eidx  = (const int*)d_in[1];
  const float* W0    = (const float*)d_in[2];
  const float* b0    = (const float*)d_in[3];
  const float* Wd    = (const float*)d_in[4];
  const float* bd    = (const float*)d_in[5];
  const float* P     = (const float*)d_in[6];
  const float* Wu    = (const float*)d_in[7];
  const float* bu    = (const float*)d_in[8];
  const float* Wlast = (const float*)d_in[9];
  const float* blast = (const float*)d_in[10];

  const int N0 = 4096;
  const int ns[5] = {4096, 2000, 1000, 500, 250};
  const int E = in_sizes[1] / 2;

  char* ws = (char*)d_ws;
  size_t off = 0;
  auto carve = [&](size_t bytes) {
    char* p = ws + off;
    off += (bytes + 255) & ~(size_t)255;
    return p;
  };
  float* Alev[4];
  Alev[0] = (float*)carve((size_t)4096 * 4096 * 4);
  Alev[1] = (float*)carve((size_t)2000 * 2000 * 4);
  Alev[2] = (float*)carve((size_t)1000 * 1000 * 4);
  Alev[3] = (float*)carve((size_t)500 * 500 * 4);
  float* A4 = (float*)carve((size_t)250 * 250 * 4);
  float* Tsq = (float*)carve((size_t)4096 * 4096 * 4);
  unsigned char* U8a = (unsigned char*)carve((size_t)4096 * 4096);
  unsigned char* U8t = (unsigned char*)carve((size_t)4096 * 4096);
  float* msgb = (float*)carve((size_t)4096 * 32 * 4);
  float* degb = (float*)carve((size_t)4096 * 4);
  float* disb = (float*)carve((size_t)4096 * 4);
  float* scb  = (float*)carve((size_t)4096 * 4);
  int*   permb[4];
  float* valsb[4];
  for (int i = 0; i < 4; ++i) { permb[i] = (int*)carve(2048 * 4); valsb[i] = (float*)carve(2048 * 4); }
  float* xs[4];
  for (int i = 0; i < 4; ++i) xs[i] = (float*)carve((size_t)4096 * 32 * 4);
  float* xbA = (float*)carve((size_t)4096 * 32 * 4);
  float* xbB = (float*)carve((size_t)4096 * 32 * 4);
  float* logits = (float*)carve((size_t)4096 * 3 * 4);

  auto cdiv = [](size_t a, size_t b) { return (int)((a + b - 1) / b); };

  auto gcn_norm = [&](const float* Am, int n) {
    k_zero_f32<<<cdiv(n, 256), 256, 0, stream>>>(degb, (size_t)n);
    const int slices = 16;
    k_colsum_part<<<cdiv((size_t)n * slices, 256), 256, 0, stream>>>(Am, n, slices, degb);
    k_finalize_dis<<<cdiv(n, 256), 256, 0, stream>>>(Am, n, degb, disb);
  };
  auto gcn32 = [&](const float* Am, int n, const float* x, int Cin,
                   const float* W, const float* b, float* out) {
    gcn_norm(Am, n);
    k_msg<<<cdiv((size_t)n * 32, 256), 256, 0, stream>>>(x, Cin, W, 32, disb, n, msgb);
    k_aggregate<32, true><<<cdiv(n, 32), 256, 0, stream>>>(Am, n, msgb, disb, b, out);
  };

  k_zero_f32<<<cdiv((size_t)N0 * N0, 256), 256, 0, stream>>>(Alev[0], (size_t)N0 * N0);
  k_build_edges<<<cdiv(E, 256), 256, 0, stream>>>(eidx, E, Alev[0], N0);

  gcn32(Alev[0], N0, x_in, 3, W0, b0, xs[0]);
  float* xcur = xs[0];

  for (int i = 0; i < 4; ++i) {
    int n = ns[i], k = ns[i + 1];
    if (i < 2) {
      int np = (n + 127) & ~127;
      dim3 bt(32, 8), gt(np / 32, np / 32);
      k_u8pad<<<gt, bt, 0, stream>>>(Alev[i], n, np, U8a, U8t);
      dim3 gg(cdiv(n, 32), cdiv(n, 64));
      k_gemm_iu8<<<gg, 256, 0, stream>>>(U8a, U8t, n, np, Tsq);
    } else {
      int np = (n + 63) & ~63;
      float* Fpad = (float*)U8a;
      k_f32pad<<<cdiv((size_t)np * np, 256), 256, 0, stream>>>(Alev[i], n, np, Fpad);
      dim3 gg(cdiv(n, 32), cdiv(n, 64));
      k_gemm_f32<<<gg, 256, 0, stream>>>(Fpad, n, np, Tsq);
    }
    k_score<<<cdiv(n, 256), 256, 0, stream>>>(xcur, P + i * 32, n, scb);
    int npad2 = 1; while (npad2 < n) npad2 <<= 1;
    k_sort_topk<<<1, 1024, 0, stream>>>(scb, n, npad2, k, permb[i], valsb[i]);
    k_gather_x<<<cdiv((size_t)k * 32, 256), 256, 0, stream>>>(xcur, permb[i], valsb[i], k, xbB);
    float* Anext = (i < 3) ? Alev[i + 1] : A4;
    k_gather_A<<<cdiv((size_t)k * k, 256), 256, 0, stream>>>(Tsq, n, permb[i], k, Anext);
    float* dest = (i < 3) ? xs[i + 1] : xbA;
    gcn32(Anext, k, xbB, 32, Wd + i * 32 * 32, bd + i * 32, dest);
    xcur = dest;
  }

  for (int i = 0; i < 4; ++i) {
    int j = 3 - i;
    int nj = ns[j], kj = ns[j + 1];
    k_copy_f32<<<cdiv((size_t)nj * 32, 256), 256, 0, stream>>>(xs[j], xbB, (size_t)nj * 32);
    k_scatter_add<<<cdiv((size_t)kj * 32, 256), 256, 0, stream>>>(xbB, permb[j], xcur, kj);
    if (i < 3) {
      gcn32(Alev[j], nj, xbB, 32, Wu + i * 32 * 32, bu + i * 32, xbA);
      xcur = xbA;
    } else {
      gcn_norm(Alev[0], N0);
      k_msg<<<cdiv((size_t)N0 * 3, 256), 256, 0, stream>>>(xbB, 32, Wlast, 3, disb, N0, msgb);
      k_aggregate<3, false><<<cdiv(N0, 32), 256, 0, stream>>>(Alev[0], N0, msgb, disb, blast, logits);
      k_logsoftmax3<<<cdiv(N0, 256), 256, 0, stream>>>(logits, N0, (float*)d_out);
    }
  }
}
